// VitDetLayer_55327768707841
// MI455X (gfx1250) — matrix operand order repaired and checked by reading, not compiled, not run on hardware
//
#include <hip/hip_runtime.h>
#include <math.h>

typedef __attribute__((ext_vector_type(16))) __bf16 v16bf;
typedef __attribute__((ext_vector_type(8)))  float  v8f;
typedef unsigned int  u32;
typedef unsigned short u16;
typedef unsigned long long u64;

union Frag { v16bf v; u32 u[8]; };

__device__ __forceinline__ u16 f2bf(float f) {
    u32 u = __builtin_bit_cast(u32, f);
    u32 r = u + 0x7fffu + ((u >> 16) & 1u);   // round-to-nearest-even
    return (u16)(r >> 16);
}
__device__ __forceinline__ float bf2f(u16 s) {
    return __builtin_bit_cast(float, (u32)s << 16);
}
// low 32 bits of a generic pointer into __shared__ = LDS byte offset (ISA 10.2)
__device__ __forceinline__ u32 lds_addr(const void* p) {
    return (u32)(size_t)p;
}

// ---------------------------------------------------------------------------
// fp32 -> bf16 weight conversion
// ---------------------------------------------------------------------------
__global__ void w2bf(const float* __restrict__ in, u16* __restrict__ out, int n) {
    int i = blockIdx.x * 256 + threadIdx.x;
    if (i < n) out[i] = f2bf(in[i]);
}

// ---------------------------------------------------------------------------
// LayerNorm1 + window partition (64x64 -> 5x5 windows of 14x14, zero pad)
// one wave per token; 39200 window-token rows, C=768
// ---------------------------------------------------------------------------
__global__ __launch_bounds__(256) void ln1_win(const float* __restrict__ x,
        const float* __restrict__ w, const float* __restrict__ b,
        u16* __restrict__ Awin) {
    int r = blockIdx.x * 8 + (threadIdx.x >> 5);
    int lane = threadIdx.x & 31;
    if (r >= 39200) return;
    int win = r / 196, n = r % 196;
    int bb = win / 25, w2 = win % 25, wy = w2 / 5, wx = w2 % 5;
    int iy = wy * 14 + n / 14, ix = wx * 14 + n % 14;
    u16* dst = Awin + (size_t)r * 768;
    if (iy >= 64 || ix >= 64) {                    // padded region -> zeros
        for (int c = lane; c < 768; c += 32) dst[c] = 0;
        return;
    }
    const float* src = x + (((size_t)bb * 64 + iy) * 64 + ix) * 768;
    float s = 0.f, s2 = 0.f;
    for (int c = lane; c < 768; c += 32) { float v = src[c]; s += v; s2 += v * v; }
    for (int m = 16; m >= 1; m >>= 1) { s += __shfl_xor(s, m, 32); s2 += __shfl_xor(s2, m, 32); }
    float mean = s * (1.f / 768.f);
    float var  = s2 * (1.f / 768.f) - mean * mean;
    float rs   = rsqrtf(var + 1e-6f);
    for (int c = lane; c < 768; c += 32)
        dst[c] = f2bf((src[c] - mean) * rs * w[c] + b[c]);
}

// LayerNorm2 over h (32768 rows) -> bf16
__global__ __launch_bounds__(256) void ln2k(const float* __restrict__ h,
        const float* __restrict__ w, const float* __restrict__ b,
        u16* __restrict__ M2) {
    int r = blockIdx.x * 8 + (threadIdx.x >> 5);
    int lane = threadIdx.x & 31;
    if (r >= 32768) return;
    const float* src = h + (size_t)r * 768;
    u16* dst = M2 + (size_t)r * 768;
    float s = 0.f, s2 = 0.f;
    for (int c = lane; c < 768; c += 32) { float v = src[c]; s += v; s2 += v * v; }
    for (int m = 16; m >= 1; m >>= 1) { s += __shfl_xor(s, m, 32); s2 += __shfl_xor(s2, m, 32); }
    float mean = s * (1.f / 768.f);
    float var  = s2 * (1.f / 768.f) - mean * mean;
    float rs   = rsqrtf(var + 1e-6f);
    for (int c = lane; c < 768; c += 32)
        dst[c] = f2bf((src[c] - mean) * rs * w[c] + b[c]);
}

// ---------------------------------------------------------------------------
// Generic bf16 WMMA GEMM: 128x128 block tile, 8 waves (2x4), each 64x32.
// A [M x K] row-major bf16, W [K x N] row-major bf16, f32 accumulate.
// A tile staged with async global->LDS DMA (ASYNCcnt); B pair-transposed
// manually (layout permute). Epilogue modes: 0 = qkv scatter,
// 1 = proj+residual scatter, 2 = fc1+GELU->bf16, 3 = fc2+residual->f32 out
// ---------------------------------------------------------------------------
__global__ __launch_bounds__(256) void gemm_bf16(
        const u16* __restrict__ A, const u16* __restrict__ W,
        const float* __restrict__ bias,
        int M, int K, int N, int mode,
        const float* __restrict__ xres, float* __restrict__ hbuf,
        u16* __restrict__ obf, float* __restrict__ fout,
        u16* __restrict__ qb, u16* __restrict__ kb, u16* __restrict__ vb) {
    __shared__ u32 As[128 * 16];   // [row][k-dword], 32 bf16 per row
    __shared__ u32 Bs[128 * 16];   // [col][k-pair dword] (pair-transposed)
    const int tid   = threadIdx.x;
    const int Mbase = blockIdx.y * 128;
    const int Nbase = blockIdx.x * 128;
    const int wave  = tid >> 5, lane = tid & 31;
    const int lo    = lane & 15, hi = lane >> 4;
    const int arow  = (wave >> 2) * 64;
    const int bcol  = (wave & 3) * 32;

    v8f acc[4][2];
    for (int i = 0; i < 4; ++i)
        for (int j = 0; j < 2; ++j)
            for (int p = 0; p < 8; ++p) acc[i][j][p] = 0.f;

    const int arow_g = tid >> 1, ahalf = tid & 1;   // A staging: 2 thr / row
    const int cg = tid & 15, kp = tid >> 4;         // B staging: 8 cols x 2 k-rows
    const int grow = Mbase + arow_g;
    const u32 a_lds = lds_addr(&As[arow_g * 16 + 8 * ahalf]);

    for (int k0 = 0; k0 < K; k0 += 32) {
        // stage A tile 128x32: async DMA global -> LDS (32B per thread);
        // rows >= M leave stale LDS that only feeds discarded accumulator rows
        if (grow < M) {
            u64 ga = (u64)(size_t)(A + (size_t)grow * K + k0 + 16 * ahalf);
            asm volatile(
                "global_load_async_to_lds_b128 %0, %1, off\n\t"
                "global_load_async_to_lds_b128 %0, %1, off offset:16"
                :: "v"(a_lds), "v"(ga) : "memory");
        }
        { // stage B tile 32x128, pair-transposed to [col][K/2]
            const uint4* w0 = reinterpret_cast<const uint4*>(
                W + (size_t)(k0 + 2 * kp) * N + Nbase + 8 * cg);
            const uint4* w1 = reinterpret_cast<const uint4*>(
                W + (size_t)(k0 + 2 * kp + 1) * N + Nbase + 8 * cg);
            uint4 r0 = w0[0], r1 = w1[0];
            u32 r0a[4] = { r0.x, r0.y, r0.z, r0.w };
            u32 r1a[4] = { r1.x, r1.y, r1.z, r1.w };
#pragma unroll
            for (int i = 0; i < 8; ++i) {
                u32 e0 = (r0a[i >> 1] >> (16 * (i & 1))) & 0xffffu;
                u32 e1 = (r1a[i >> 1] >> (16 * (i & 1))) & 0xffffu;
                Bs[(8 * cg + i) * 16 + kp] = e0 | (e1 << 16);
            }
        }
        asm volatile("s_wait_asynccnt 0x0" ::: "memory");
        __syncthreads();
        Frag af[4], bfr[2];
#pragma unroll
        for (int i = 0; i < 4; ++i) {
            const u32* p = &As[(arow + 16 * i + lo) * 16];
#pragma unroll
            for (int v = 0; v < 4; ++v) {
                af[i].u[v]     = p[4 * hi + v];
                af[i].u[4 + v] = p[8 + 4 * hi + v];
            }
        }
#pragma unroll
        for (int j = 0; j < 2; ++j) {
            const u32* p = &Bs[(bcol + 16 * j + lo) * 16];
#pragma unroll
            for (int v = 0; v < 4; ++v) { bfr[j].u[v] = p[4 * hi + v]; bfr[j].u[4 + v] = p[8 + 4 * hi + v]; }
        }
#pragma unroll
        for (int i = 0; i < 4; ++i)
#pragma unroll
            for (int j = 0; j < 2; ++j)
                acc[i][j] = __builtin_amdgcn_wmma_f32_16x16x32_bf16(
                    false, af[i].v, false, bfr[j].v, (short)0, acc[i][j], false, false);
        __syncthreads();
    }

    // epilogue
    for (int i = 0; i < 4; ++i)
        for (int j = 0; j < 2; ++j)
            for (int p = 0; p < 8; ++p) {
                int row = Mbase + arow + 16 * i + p + 8 * hi;
                int col = Nbase + bcol + 16 * j + lo;
                if (row >= M) continue;
                float v = acc[i][j][p];
                if (mode == 0) {            // qkv -> per-head bf16 scatter (q unscaled)
                    float val = v + bias[col];
                    int win = row / 196, n = row % 196;
                    int three = col / 768, cw = col % 768;
                    int head = cw >> 6, d = cw & 63;
                    size_t idx = (((size_t)win * 12 + head) * 196 + n) * 64 + d;
                    u16 bv = f2bf(val);
                    if (three == 0) qb[idx] = bv;
                    else if (three == 1) kb[idx] = bv;
                    else vb[idx] = bv;
                } else if (mode == 1) {     // proj + window-unpartition + residual
                    float val = v + bias[col];
                    int win = row / 196, n = row % 196;
                    int bb = win / 25, w2 = win % 25, wy = w2 / 5, wx = w2 % 5;
                    int iy = wy * 14 + n / 14, ix = wx * 14 + n % 14;
                    if (iy < 64 && ix < 64) {
                        size_t g = (((size_t)bb * 64 + iy) * 64 + ix) * 768 + col;
                        hbuf[g] = xres[g] + val;
                    }
                } else if (mode == 2) {     // fc1 + exact GELU -> bf16
                    float val = v + bias[col];
                    float gg = 0.5f * val * (1.f + erff(val * 0.70710678118f));
                    obf[(size_t)row * N + col] = f2bf(gg);
                } else {                    // fc2 + residual -> f32 output
                    size_t g = (size_t)row * 768 + col;
                    fout[g] = hbuf[g] + v + bias[col];
                }
            }
}

// ---------------------------------------------------------------------------
// Decomposed rel-pos bias terms: relh[hw][i][k] = q[hw][i,:] . Rh[i/14 - k + 13,:]
//                                relw[hw][i][l] = q[hw][i,:] . Rw[i%14 - l + 13,:]
// ---------------------------------------------------------------------------
__global__ __launch_bounds__(256) void relpos(const u16* __restrict__ qb,
        const float* __restrict__ rph, const float* __restrict__ rpw,
        float* __restrict__ relh, float* __restrict__ relw) {
    int idx = blockIdx.x * 256 + threadIdx.x;
    if (idx >= 2400 * 196 * 14) return;
    int k  = idx % 14;
    int t  = idx / 14;
    int i  = t % 196;
    int hw = t / 196;
    int a = i / 14, ww = i % 14;
    const u16*   q  = qb  + ((size_t)hw * 196 + i) * 64;
    const float* rh = rph + (size_t)(a - k + 13) * 64;
    const float* rw = rpw + (size_t)(ww - k + 13) * 64;
    float sh = 0.f, sw = 0.f;
    for (int d = 0; d < 64; ++d) {
        float qv = bf2f(q[d]);
        sh += qv * rh[d];
        sw += qv * rw[d];
    }
    relh[idx] = sh;
    relw[idx] = sw;
}

// ---------------------------------------------------------------------------
// Attention: one block (7 waves, 224 threads) per (window, head).
// N padded 196 -> 224. S = q k^T via WMMA, +bias, softmax, O = P v via WMMA.
// K staged via async global->LDS DMA; V^T staged manually (transpose permute).
// ---------------------------------------------------------------------------
__global__ __launch_bounds__(224) void attn(
        const u16* __restrict__ qb, const u16* __restrict__ kb,
        const u16* __restrict__ vb,
        const float* __restrict__ relh, const float* __restrict__ relw,
        u16* __restrict__ obuf) {
    __shared__ u32 kS[224 * 32];     // K  [token][d/2]            28672 B
    __shared__ u32 vtS[64 * 112];    // V^T [d][token-pair]        28672 B
    __shared__ u32 pS[7 * 256];      // per-wave P tile 16x32 bf16  7168 B
    int hw = blockIdx.x;
    int win = hw / 12, head = hw % 12;
    int tid = threadIdx.x;
    { // stage K (async DMA; rows >= 196 stale -> masked to -inf later) and V^T
        int j = tid;
        bool valid = j < 196;
        if (valid) {
            u32 klds = lds_addr(&kS[j * 32]);
            u64 ga = (u64)(size_t)(kb + ((size_t)hw * 196 + j) * 64);
            asm volatile(
                "global_load_async_to_lds_b128 %0, %1, off\n\t"
                "global_load_async_to_lds_b128 %0, %1, off offset:16\n\t"
                "global_load_async_to_lds_b128 %0, %1, off offset:32\n\t"
                "global_load_async_to_lds_b128 %0, %1, off offset:48"
                :: "v"(klds), "v"(ga) : "memory");
        }
        const u32* vr = reinterpret_cast<const u32*>(vb + ((size_t)hw * 196 + (valid ? j : 0)) * 64);
        u16* vt = reinterpret_cast<u16*>(vtS);
        for (int dw = 0; dw < 32; ++dw) {
            u32 vv = valid ? vr[dw] : 0u;   // V must be true zeros (P*V sum)
            vt[(2 * dw) * 224 + j]     = (u16)(vv & 0xffffu);
            vt[(2 * dw + 1) * 224 + j] = (u16)(vv >> 16);
        }
    }
    asm volatile("s_wait_asynccnt 0x0" ::: "memory");
    __syncthreads();
    int wave = tid >> 5, lane = tid & 31, lo = lane & 15, hi = lane >> 4;
    u16* pw = reinterpret_cast<u16*>(pS) + wave * 512;
    const u32* pr = &pS[wave * 256];

    for (int sidx = 0; sidx < 2; ++sidx) {
        int s = wave * 2 + sidx;
        int rowbase = 16 * s;
        // q A-fragments directly from global (rows >= 196 -> zero)
        Frag qa[2];
        {
            int row = rowbase + lo;
            bool okr = row < 196;
            const u32* qp = reinterpret_cast<const u32*>(
                qb + ((size_t)hw * 196 + (okr ? row : 0)) * 64);
            for (int kt = 0; kt < 2; ++kt)
                for (int v = 0; v < 4; ++v) {
                    qa[kt].u[v]     = okr ? qp[16 * kt + 4 * hi + v] : 0u;
                    qa[kt].u[4 + v] = okr ? qp[16 * kt + 8 + 4 * hi + v] : 0u;
                }
        }
        // S = q k^T
        v8f sc[14];
        for (int t = 0; t < 14; ++t) for (int p = 0; p < 8; ++p) sc[t][p] = 0.f;
        for (int t = 0; t < 14; ++t) {
            const u32* kp2 = &kS[(16 * t + lo) * 32];
            Frag b0, b1;
            for (int v = 0; v < 4; ++v) { b0.u[v] = kp2[4 * hi + v]; b0.u[4 + v] = kp2[8 + 4 * hi + v]; b1.u[v] = kp2[16 + 4 * hi + v]; b1.u[4 + v] = kp2[24 + 4 * hi + v]; }
            sc[t] = __builtin_amdgcn_wmma_f32_16x16x32_bf16(false, qa[0].v, false, b0.v, (short)0, sc[t], false, false);
            sc[t] = __builtin_amdgcn_wmma_f32_16x16x32_bf16(false, qa[1].v, false, b1.v, (short)0, sc[t], false, false);
        }
        // scale + rel-pos bias + pad mask (overwrites, so stale K rows vanish)
        for (int t = 0; t < 14; ++t)
            for (int p = 0; p < 8; ++p) {
                int i = rowbase + p + 8 * hi;
                int j = 16 * t + lo;
                float val;
                if (i < 196 && j < 196) {
                    size_t rb = ((size_t)hw * 196 + i) * 14;
                    val = sc[t][p] * 0.125f + relh[rb + j / 14] + relw[rb + j % 14];
                } else val = -1e30f;
                sc[t][p] = val;
            }
        // softmax per row (rows live in 16-lane half-groups)
        for (int p = 0; p < 8; ++p) {
            float m = -1e30f;
            for (int t = 0; t < 14; ++t) m = fmaxf(m, sc[t][p]);
            for (int msk = 8; msk >= 1; msk >>= 1) m = fmaxf(m, __shfl_xor(m, msk, 32));
            float ssum = 0.f;
            for (int t = 0; t < 14; ++t) { float e = __expf(sc[t][p] - m); sc[t][p] = e; ssum += e; }
            for (int msk = 8; msk >= 1; msk >>= 1) ssum += __shfl_xor(ssum, msk, 32);
            float inv = 1.f / ssum;
            for (int t = 0; t < 14; ++t) sc[t][p] *= inv;
        }
        // O = P @ V  (restage P through LDS to get the A-operand layout)
        v8f oacc[4];
        for (int j = 0; j < 4; ++j) for (int p = 0; p < 8; ++p) oacc[j][p] = 0.f;
        for (int kt = 0; kt < 7; ++kt) {
            for (int tt = 0; tt < 2; ++tt) {
                int t = 2 * kt + tt;
                for (int p = 0; p < 8; ++p) {
                    int rl = p + 8 * hi, cl = 16 * tt + lo;
                    pw[rl * 32 + cl] = f2bf(sc[t][p]);
                }
            }
            asm volatile("s_wait_dscnt 0x0" ::: "memory"); // P stores visible before reload
            Frag pa;
            {
                const u32* pp = pr + lo * 16;
                for (int v = 0; v < 8; ++v) pa.u[v] = pp[8 * hi + v];
            }
            for (int j = 0; j < 4; ++j) {
                Frag vf;
                int c = 16 * j + lo;
                const u32* vp = &vtS[c * 112 + 16 * kt + 8 * hi];
                for (int v = 0; v < 8; ++v) vf.u[v] = vp[v];
                oacc[j] = __builtin_amdgcn_wmma_f32_16x16x32_bf16(false, pa.v, false, vf.v, (short)0, oacc[j], false, false);
            }
        }
        // store O (bf16, head-major columns) for proj GEMM
        for (int j = 0; j < 4; ++j)
            for (int p = 0; p < 8; ++p) {
                int i = rowbase + p + 8 * hi;
                if (i < 196) {
                    int d = 16 * j + lo;
                    obuf[((size_t)win * 196 + i) * 768 + head * 64 + d] = f2bf(oacc[j][p]);
                }
            }
    }
}

// ---------------------------------------------------------------------------
extern "C" void kernel_launch(void* const* d_in, const int* in_sizes, int n_in,
                              void* d_out, int out_size, void* d_ws, size_t ws_size,
                              hipStream_t stream) {
    const float* x     = (const float*)d_in[0];
    const float* ln1w  = (const float*)d_in[1];
    const float* ln1b  = (const float*)d_in[2];
    const float* qkvw  = (const float*)d_in[3];
    const float* qkvb  = (const float*)d_in[4];
    const float* rph   = (const float*)d_in[5];
    const float* rpw   = (const float*)d_in[6];
    const float* projw = (const float*)d_in[7];
    const float* projb = (const float*)d_in[8];
    const float* ln2w  = (const float*)d_in[9];
    const float* ln2b  = (const float*)d_in[10];
    const float* fc1w  = (const float*)d_in[11];
    const float* fc1b  = (const float*)d_in[12];
    const float* fc2w  = (const float*)d_in[13];
    const float* fc2b  = (const float*)d_in[14];
    float* out = (float*)d_out;

    char* base = (char*)d_ws;
    size_t off = 0;
    auto alloc = [&](size_t bytes) -> void* {
        void* p = base + off;
        off = (off + bytes + 255) & ~(size_t)255;
        return p;
    };
    u16*  qkvwb  = (u16*)alloc((size_t)768 * 2304 * 2);
    u16*  projwb = (u16*)alloc((size_t)768 * 768 * 2);
    u16*  fc1wb  = (u16*)alloc((size_t)768 * 3072 * 2);
    u16*  fc2wb  = (u16*)alloc((size_t)3072 * 768 * 2);
    u16*  Awin   = (u16*)alloc((size_t)39200 * 768 * 2);
    u16*  qB     = (u16*)alloc((size_t)2400 * 196 * 64 * 2);
    u16*  kB     = (u16*)alloc((size_t)2400 * 196 * 64 * 2);
    u16*  vB     = (u16*)alloc((size_t)2400 * 196 * 64 * 2);
    float* relH  = (float*)alloc((size_t)2400 * 196 * 14 * 4);
    float* relW  = (float*)alloc((size_t)2400 * 196 * 14 * 4);
    u16*  Obuf   = (u16*)alloc((size_t)39200 * 768 * 2);
    float* hbuf  = (float*)alloc((size_t)32768 * 768 * 4);
    u16*  M2     = (u16*)alloc((size_t)32768 * 768 * 2);
    u16*  Gbuf   = (u16*)alloc((size_t)32768 * 3072 * 2);

    w2bf<<<(1769472 + 255) / 256, 256, 0, stream>>>(qkvw, qkvwb, 1769472);
    w2bf<<<(589824  + 255) / 256, 256, 0, stream>>>(projw, projwb, 589824);
    w2bf<<<(2359296 + 255) / 256, 256, 0, stream>>>(fc1w, fc1wb, 2359296);
    w2bf<<<(2359296 + 255) / 256, 256, 0, stream>>>(fc2w, fc2wb, 2359296);

    ln1_win<<<4900, 256, 0, stream>>>(x, ln1w, ln1b, Awin);

    gemm_bf16<<<dim3(18, 307), 256, 0, stream>>>(Awin, qkvwb, qkvb,
        39200, 768, 2304, 0, nullptr, nullptr, nullptr, nullptr, qB, kB, vB);

    relpos<<<(2400 * 196 * 14 + 255) / 256, 256, 0, stream>>>(qB, rph, rpw, relH, relW);

    attn<<<2400, 224, 0, stream>>>(qB, kB, vB, relH, relW, Obuf);

    gemm_bf16<<<dim3(6, 307), 256, 0, stream>>>(Obuf, projwb, projb,
        39200, 768, 768, 1, x, hbuf, nullptr, nullptr, nullptr, nullptr, nullptr);

    ln2k<<<4096, 256, 0, stream>>>(hbuf, ln2w, ln2b, M2);

    gemm_bf16<<<dim3(24, 256), 256, 0, stream>>>(M2, fc1wb, fc1b,
        32768, 768, 3072, 2, nullptr, nullptr, Gbuf, nullptr, nullptr, nullptr, nullptr);

    gemm_bf16<<<dim3(6, 256), 256, 0, stream>>>(Gbuf, fc2wb, fc2b,
        32768, 3072, 768, 3, nullptr, hbuf, nullptr, out, nullptr, nullptr, nullptr);
}
